// DTFBlock_19069654794409
// MI455X (gfx1250) — matrix operand order repaired and checked by reading, not compiled, not run on hardware
//
#include <hip/hip_runtime.h>

// ---------------- problem constants ----------------
constexpr int Bc  = 4;
constexpr int Cc  = 128;
constexpr int Tc  = 16;
constexpr int Hc  = 56;
constexpr int Wc  = 56;
constexpr int KKc = 9;            // 3x3 taps
constexpr int tc  = 9;            // T/2+1 rfft bins
constexpr int HWc = Hc * Wc;      // 3136
constexpr int Kc_ = (Cc + KKc) * Tc;   // 2192  (GEMM1 K)
constexpr int Kp  = 2208;              // K padded to multiple of 32
constexpr int M1  = Cc * tc * 2 / 16;  // 144   (GEMM1 M)
constexpr int KT1 = Kp / 32;           // 69 k-tiles (GEMM1)
constexpr int KT2 = Cc / 32;           // 4  k-tiles (GEMM2)
constexpr int NT  = HWc / 16;          // 196 n-tiles
constexpr int TILE = 512;              // f16 elems per 16x32 / 32x16 tile

typedef __attribute__((ext_vector_type(16))) _Float16 v16h;
typedef __attribute__((ext_vector_type(8)))  float    v8f;

// cos/sin(j*pi/8), j = 0..15  (all DFT phases are (k*n) mod 16)
__constant__ float COS16[16] = {
     1.f,  0.92387953f,  0.70710678f,  0.38268343f,  0.f, -0.38268343f,
    -0.70710678f, -0.92387953f, -1.f, -0.92387953f, -0.70710678f,
    -0.38268343f,  0.f,  0.38268343f,  0.70710678f,  0.92387953f };
__constant__ float SIN16[16] = {
     0.f,  0.38268343f,  0.70710678f,  0.92387953f,  1.f,  0.92387953f,
     0.70710678f,  0.38268343f,  0.f, -0.38268343f, -0.70710678f,
    -0.92387953f, -1.f, -0.92387953f, -0.70710678f, -0.38268343f };

// ---- WMMA fragment swizzle maps (CDNA5 16x16x32 f16 VGPR layouts) ------
// A 16x32 (m,k) -> offset inside 512-elem tile
__device__ __forceinline__ int a_swizzle(int mloc, int kloc) {
  int half, e;
  if (kloc < 16) { half = kloc >> 3; e = kloc & 7; }
  else           { half = (kloc - 16) >> 3; e = 8 + ((kloc - 16) & 7); }
  return (half * 16 + mloc) * 16 + e;
}
// B 32x16 (k,n) -> offset inside 512-elem tile
__device__ __forceinline__ int b_swizzle(int kloc, int nloc) {
  return (((kloc >> 3) & 1) * 16 + nloc) * 16 + ((kloc >> 4) << 3) + (kloc & 7);
}
// fragment load: 16 contiguous f16 per lane (one 32B load)
__device__ __forceinline__ v16h load_frag(const _Float16* __restrict__ tile,
                                          int lane) {
  return *(const v16h*)(tile + lane * 16);
}

// ---------------- K0: 3x3 temporal correlation w_o ----------------
__global__ __launch_bounds__(256) void corr_kernel(
    const float* __restrict__ x, const float* __restrict__ wcor,
    float* __restrict__ w_o) {
  int idx = blockIdx.x * 256 + threadIdx.x;
  if (idx >= Bc * Tc * HWc) return;
  int hw = idx % HWc;
  int tt = (idx / HWc) % Tc;
  int b  = idx / (HWc * Tc);
  int h = hw / Wc, w = hw % Wc;
  int tn = (tt + 1 < Tc) ? tt + 1 : Tc - 1;
  float acc[KKc];
#pragma unroll
  for (int k = 0; k < KKc; ++k) acc[k] = 0.f;
  for (int c = 0; c < Cc; ++c) {
    const float* xc = x + ((size_t)(b * Cc + c) * Tc) * HWc;
    float xv = xc[tt * HWc + hw];
    const float* xn = xc + tn * HWc;
    const float* wc = wcor + (c * Tc + tt) * KKc;
#pragma unroll
    for (int ki = 0; ki < 3; ++ki) {
      int hh = h + ki - 1;
      if (hh < 0 || hh >= Hc) continue;
#pragma unroll
      for (int kj = 0; kj < 3; ++kj) {
        int ww = w + kj - 1;
        if (ww < 0 || ww >= Wc) continue;
        acc[ki * 3 + kj] = fmaf(xv * xn[hh * Wc + ww], wc[ki * 3 + kj],
                                acc[ki * 3 + kj]);
      }
    }
  }
#pragma unroll
  for (int k = 0; k < KKc; ++k)
    w_o[((size_t)(b * KKc + k) * Tc + tt) * HWc + hw] = acc[k];
}

// ---------------- K1: softmax taps + aggregate + temporal shift -> x2 ----
__global__ __launch_bounds__(256) void agg_kernel(
    const float* __restrict__ x, const float* __restrict__ w_o,
    float* __restrict__ x2) {
  int idx = blockIdx.x * 256 + threadIdx.x;
  if (idx >= Bc * Tc * HWc) return;
  int hw = idx % HWc;
  int tt = (idx / HWc) % Tc;
  int b  = idx / (HWc * Tc);
  int h = hw / Wc, w = hw % Wc;

  if (tt == Tc - 1) {  // a_shift is zero on the last frame
    for (int c = 0; c < Cc; ++c) {
      size_t i = ((size_t)(b * Cc + c) * Tc + tt) * HWc + hw;
      x2[i] = x[i];
    }
    return;
  }
  int ts = tt + 1;
  float ws[KKc];
  float m = -1e30f;
#pragma unroll
  for (int k = 0; k < KKc; ++k) {
    ws[k] = w_o[((size_t)(b * KKc + k) * Tc + ts) * HWc + hw];
    m = fmaxf(m, ws[k]);
  }
  float s = 0.f;
#pragma unroll
  for (int k = 0; k < KKc; ++k) { ws[k] = __expf(ws[k] - m); s += ws[k]; }
  float inv = 1.f / s;
#pragma unroll
  for (int k = 0; k < KKc; ++k) ws[k] *= inv;

  for (int c = 0; c < Cc; ++c) {
    const float* xc = x + ((size_t)(b * Cc + c) * Tc) * HWc;
    const float* xs = xc + ts * HWc;
    float acc = 0.f;
#pragma unroll
    for (int ki = 0; ki < 3; ++ki) {
      int hh = h + ki - 1;
      if (hh < 0 || hh >= Hc) continue;
#pragma unroll
      for (int kj = 0; kj < 3; ++kj) {
        int ww = w + kj - 1;
        if (ww < 0 || ww >= Wc) continue;
        acc = fmaf(ws[ki * 3 + kj], xs[hh * Wc + ww], acc);
      }
    }
    x2[((size_t)(b * Cc + c) * Tc + tt) * HWc + hw] = xc[tt * HWc + hw] + acc;
  }
}

// ---------------- K2: pack f16 operands into WMMA tile layout ------------
// Xc tiles: [b][ktile 0..68][ntile 0..195] x 512
__global__ __launch_bounds__(256) void pack_xc_kernel(
    const float* __restrict__ x2, const float* __restrict__ w_o,
    _Float16* __restrict__ Xc) {
  int idx = blockIdx.x * 256 + threadIdx.x;
  if (idx >= Bc * Kp * HWc) return;
  int hw = idx % HWc;
  int k  = (idx / HWc) % Kp;
  int b  = idx / (HWc * Kp);
  float v = 0.f;
  if (k < Kc_) {
    int cc = k >> 4;        // k = cc*T + tt, T == 16
    int tt = k & 15;
    v = (cc < Cc)
            ? x2[((size_t)(b * Cc + cc) * Tc + tt) * HWc + hw]
            : w_o[((size_t)(b * KKc + (cc - Cc)) * Tc + tt) * HWc + hw];
  }
  size_t dst = (((size_t)b * KT1 + (k >> 5)) * NT + (hw >> 4)) * TILE +
               b_swizzle(k & 31, hw & 15);
  Xc[dst] = (_Float16)v;
}

// Wg tiles: [mtile 0..8][ktile 0..68] x 512
__global__ __launch_bounds__(256) void pack_wg_kernel(
    const float* __restrict__ Wg, _Float16* __restrict__ A1) {
  int idx = blockIdx.x * 256 + threadIdx.x;
  if (idx >= M1 * Kp) return;
  int k = idx % Kp, m = idx / Kp;
  float v = (k < Kc_) ? Wg[m * Kc_ + k] : 0.f;
  size_t dst = ((size_t)(m >> 4) * KT1 + (k >> 5)) * TILE +
               a_swizzle(m & 15, k & 31);
  A1[dst] = (_Float16)v;
}

// Wl tiles: [mtile 0..7][ktile 0..3] x 512, three variants
__global__ __launch_bounds__(256) void pack_wl_kernel(
    const float* __restrict__ Wlr, const float* __restrict__ Wli,
    _Float16* __restrict__ aR, _Float16* __restrict__ aI,
    _Float16* __restrict__ aIn) {
  int idx = blockIdx.x * 256 + threadIdx.x;
  if (idx >= Cc * Cc) return;
  int k = idx % Cc, m = idx / Cc;
  size_t dst = ((size_t)(m >> 4) * KT2 + (k >> 5)) * TILE +
               a_swizzle(m & 15, k & 31);
  aR[dst]  = (_Float16)Wlr[idx];
  aI[dst]  = (_Float16)Wli[idx];
  aIn[dst] = (_Float16)(-Wli[idx]);  // f16 WMMA NEG bits must be 0 -> pre-negate
}

// ---------------- K3: GEMM1  iterm = Wg x x_c  (WMMA f16->f32) ----------
__global__ __launch_bounds__(128) void gemm1_kernel(
    const _Float16* __restrict__ A1, const _Float16* __restrict__ Xc,
    float* __restrict__ iterm) {
  int lane = threadIdx.x & 31, wave = threadIdx.x >> 5;
  int b  = blockIdx.z;
  int mt = blockIdx.y;
  int nt = blockIdx.x * 4 + wave;
  const _Float16* At = A1 + (size_t)mt * KT1 * TILE;
  const _Float16* Bt = Xc + ((size_t)b * KT1 * NT + nt) * TILE;
  union { v8f v; float e[8]; } acc;
#pragma unroll
  for (int r = 0; r < 8; ++r) acc.e[r] = 0.f;
  for (int kt = 0; kt < KT1; ++kt) {
    __builtin_prefetch(At + (kt + 1) * TILE + lane * 16, 0, 1);
    __builtin_prefetch(Bt + (size_t)(kt + 1) * NT * TILE + lane * 16, 0, 1);
    v16h a  = load_frag(At + kt * TILE, lane);
    v16h bb = load_frag(Bt + (size_t)kt * NT * TILE, lane);
    acc.v = __builtin_amdgcn_wmma_f32_16x16x32_f16(false, a, false, bb,
                                                   (short)0, acc.v, false,
                                                   false);
  }
  int half = lane >> 4, col = lane & 15;
#pragma unroll
  for (int r = 0; r < 8; ++r) {
    int m = mt * 16 + r + half * 8;
    iterm[((size_t)b * M1 + m) * HWc + nt * 16 + col] = acc.e[r];
  }
}

// ---------------- K4: 16-pt rDFT + complex gating -> swizzled f16 y ------
// ybf tiles: [b][tk][plane r/i][ktile 0..3][ntile 0..195] x 512
__global__ __launch_bounds__(256) void dft_mod_kernel(
    const float* __restrict__ x2, const float* __restrict__ iterm,
    const float* __restrict__ w1r, const float* __restrict__ w1i,
    _Float16* __restrict__ ybf) {
  int idx = blockIdx.x * 256 + threadIdx.x;
  if (idx >= Bc * Cc * HWc) return;
  int hw = idx % HWc;
  int c  = (idx / HWc) % Cc;
  int b  = idx / (HWc * Cc);
  float xv[Tc];
  const float* xp = x2 + ((size_t)(b * Cc + c) * Tc) * HWc + hw;
#pragma unroll
  for (int n = 0; n < Tc; ++n) xv[n] = xp[n * HWc];

  const int sw = ((size_t)(c >> 5) * NT + (hw >> 4)) * TILE +
                 b_swizzle(c & 31, hw & 15);
  const size_t PLANE = (size_t)KT2 * NT * TILE;

#pragma unroll
  for (int tk = 0; tk < tc; ++tk) {
    float xr = 0.f, xi = 0.f;
#pragma unroll
    for (int n = 0; n < Tc; ++n) {
      int ph = (tk * n) & 15;
      xr = fmaf(xv[n], COS16[ph], xr);
      xi = fmaf(-xv[n], SIN16[ph], xi);
    }
    xr *= 0.25f; xi *= 0.25f;   // ortho: 1/sqrt(16)
    // repeat(G)+reshape makes real/imag gates identical: ffc = g*(1+i)
    float g  = iterm[((size_t)b * M1 + ((c * 18 + 2 * tk) >> 4)) * HWc + hw];
    float ar = (xr - xi) * g, ai = (xr + xi) * g;
    float wr = w1r[c * tc + tk], wi = w1i[c * tc + tk];
    float yr = ar * wr - ai * wi, yi = ar * wi + ai * wr;
    size_t base = (size_t)((b * tc + tk) * 2) * PLANE + sw;
    ybf[base]         = (_Float16)yr;
    ybf[base + PLANE] = (_Float16)yi;
  }
}

// ---------------- K5: GEMM2  z = Wl (complex) x y  (WMMA) ---------------
__global__ __launch_bounds__(128) void gemm2_kernel(
    const _Float16* __restrict__ aRm, const _Float16* __restrict__ aIm,
    const _Float16* __restrict__ aInm, const _Float16* __restrict__ ybf,
    float* __restrict__ zr, float* __restrict__ zi) {
  int lane = threadIdx.x & 31, wave = threadIdx.x >> 5;
  int bt = blockIdx.z;
  int b  = bt / tc, tk = bt - b * tc;
  int mt = blockIdx.y;
  int nt = blockIdx.x * 4 + wave;
  const size_t PLANE = (size_t)KT2 * NT * TILE;
  const _Float16* Br = ybf + (size_t)((b * tc + tk) * 2) * PLANE + (size_t)nt * TILE;
  const _Float16* Bi = Br + PLANE;
  const _Float16* Ar = aRm  + (size_t)mt * KT2 * TILE;
  const _Float16* Ai = aIm  + (size_t)mt * KT2 * TILE;
  const _Float16* An = aInm + (size_t)mt * KT2 * TILE;
  union { v8f v; float e[8]; } accR, accI;
#pragma unroll
  for (int r = 0; r < 8; ++r) { accR.e[r] = 0.f; accI.e[r] = 0.f; }
#pragma unroll
  for (int kt = 0; kt < KT2; ++kt) {
    __builtin_prefetch(Br + (size_t)(kt + 1) * NT * TILE + lane * 16, 0, 1);
    __builtin_prefetch(Bi + (size_t)(kt + 1) * NT * TILE + lane * 16, 0, 1);
    v16h ar  = load_frag(Ar + kt * TILE, lane);
    v16h ai  = load_frag(Ai + kt * TILE, lane);
    v16h ain = load_frag(An + kt * TILE, lane);
    v16h br  = load_frag(Br + (size_t)kt * NT * TILE, lane);
    v16h bi  = load_frag(Bi + (size_t)kt * NT * TILE, lane);
    // zr = Wlr*yr - Wli*yi ; zi = Wlr*yi + Wli*yr
    accR.v = __builtin_amdgcn_wmma_f32_16x16x32_f16(false, ar, false, br,
                                                    (short)0, accR.v, false, false);
    accR.v = __builtin_amdgcn_wmma_f32_16x16x32_f16(false, ain, false, bi,
                                                    (short)0, accR.v, false, false);
    accI.v = __builtin_amdgcn_wmma_f32_16x16x32_f16(false, ar, false, bi,
                                                    (short)0, accI.v, false, false);
    accI.v = __builtin_amdgcn_wmma_f32_16x16x32_f16(false, ai, false, br,
                                                    (short)0, accI.v, false, false);
  }
  int half = lane >> 4, col = lane & 15;
#pragma unroll
  for (int r = 0; r < 8; ++r) {
    int m = mt * 16 + r + half * 8;
    size_t o = ((size_t)(b * Cc + m) * tc + tk) * HWc + nt * 16 + col;
    zr[o] = accR.e[r];
    zi[o] = accI.e[r];
  }
}

// ---------------- K6: BN statistics ----------------
__global__ __launch_bounds__(256) void zero_stats_kernel(float* __restrict__ s) {
  int i = blockIdx.x * 256 + threadIdx.x;
  if (i < 4 * Cc) s[i] = 0.f;
}

__global__ __launch_bounds__(256) void stats_kernel(
    const float* __restrict__ zr, const float* __restrict__ zi,
    float* __restrict__ stats) {
  int c = blockIdx.x, b = blockIdx.y, plane = blockIdx.z;
  const float* src = (plane == 0 ? zr : zi) + ((size_t)(b * Cc + c) * tc) * HWc;
  float s = 0.f, s2 = 0.f;
  for (int i = threadIdx.x; i < tc * HWc; i += 256) {
    float v = src[i];
    s += v; s2 = fmaf(v, v, s2);
  }
  __shared__ float sh[256], sh2[256];
  int tid = threadIdx.x;
  sh[tid] = s; sh2[tid] = s2;
  __syncthreads();
  for (int off = 128; off > 0; off >>= 1) {
    if (tid < off) { sh[tid] += sh[tid + off]; sh2[tid] += sh2[tid + off]; }
    __syncthreads();
  }
  if (tid == 0) {
    atomicAdd(&stats[plane * 2 * Cc + c], sh[0]);
    atomicAdd(&stats[plane * 2 * Cc + Cc + c], sh2[0]);
  }
}

// ---------------- K7: BN apply + irDFT + residual ----------------
__global__ __launch_bounds__(256) void final_kernel(
    const float* __restrict__ x, const float* __restrict__ zr,
    const float* __restrict__ zi, const float* __restrict__ stats,
    const float* __restrict__ gamma_r, const float* __restrict__ beta_r,
    const float* __restrict__ gamma_i, const float* __restrict__ beta_i,
    const float* __restrict__ alpha1, const float* __restrict__ bias_p,
    float* __restrict__ out) {
  int idx = blockIdx.x * 256 + threadIdx.x;
  if (idx >= Bc * Cc * HWc) return;
  int hw = idx % HWc;
  int c  = (idx / HWc) % Cc;
  int b  = idx / (HWc * Cc);

  const float invN = 1.f / (float)(Bc * tc * HWc);
  float mr = stats[c] * invN;
  float vr = stats[Cc + c] * invN - mr * mr;
  float mi = stats[2 * Cc + c] * invN;
  float vi = stats[3 * Cc + c] * invN - mi * mi;
  float sr = gamma_r[c] * rsqrtf(vr + 1e-5f);
  float br = beta_r[c] - mr * sr;
  float si = gamma_i[c] * rsqrtf(vi + 1e-5f);
  float bi = beta_i[c] - mi * si;
  float al = alpha1[c], bp = bias_p[c];

  float Zr[tc], Zi[tc];
  size_t zbase = ((size_t)(b * Cc + c) * tc) * HWc + hw;
#pragma unroll
  for (int tk = 0; tk < tc; ++tk) {
    Zr[tk] = (zr[zbase + tk * HWc] * sr + br) * al;
    Zi[tk] = (zi[zbase + tk * HWc] * si + bi) * al;
  }
  const float* xb = x + ((size_t)(b * Cc + c) * Tc) * HWc + hw;
  float* ob = out + ((size_t)(b * Cc + c) * Tc) * HWc + hw;
#pragma unroll
  for (int n = 0; n < Tc; ++n) {
    float v = Zr[0] + Zr[8] * COS16[(8 * n) & 15];
#pragma unroll
    for (int k = 1; k < 8; ++k) {
      int ph = (k * n) & 15;
      v += 2.f * (Zr[k] * COS16[ph] - Zi[k] * SIN16[ph]);
    }
    ob[n * HWc] = xb[n * HWc] + 0.25f * v + bp;  // ortho irfft: 1/sqrt(16)
  }
}

// ---------------- host ----------------
extern "C" void kernel_launch(void* const* d_in, const int* in_sizes, int n_in,
                              void* d_out, int out_size, void* d_ws,
                              size_t ws_size, hipStream_t stream) {
  (void)in_sizes; (void)n_in; (void)out_size; (void)ws_size;
  const float* x       = (const float*)d_in[0];
  const float* wcor    = (const float*)d_in[1];
  const float* Wg      = (const float*)d_in[2];
  const float* w1r     = (const float*)d_in[3];
  const float* w1i     = (const float*)d_in[4];
  const float* Wlr     = (const float*)d_in[5];
  const float* Wli     = (const float*)d_in[6];
  const float* gamma_r = (const float*)d_in[7];
  const float* beta_r  = (const float*)d_in[8];
  const float* gamma_i = (const float*)d_in[9];
  const float* beta_i  = (const float*)d_in[10];
  const float* alpha1  = (const float*)d_in[11];
  const float* bias_p  = (const float*)d_in[12];
  float* out = (float*)d_out;

  char* ws = (char*)d_ws;
  size_t off = 0;
  auto alloc = [&](size_t bytes) -> void* {
    void* p = ws + off;
    off += (bytes + 255) & ~(size_t)255;
    return p;
  };
  float*    w_o   = (float*)alloc(sizeof(float) * (size_t)Bc * KKc * Tc * HWc);
  float*    x2    = (float*)alloc(sizeof(float) * (size_t)Bc * Cc * Tc * HWc);
  _Float16* A1    = (_Float16*)alloc(sizeof(_Float16) * (size_t)M1 * Kp);
  _Float16* Xc    = (_Float16*)alloc(sizeof(_Float16) * (size_t)Bc * Kp * HWc);
  float*    iterm = (float*)alloc(sizeof(float) * (size_t)Bc * M1 * HWc);
  _Float16* ybf   = (_Float16*)alloc(sizeof(_Float16) * (size_t)Bc * tc * 2 * Cc * HWc);
  _Float16* aR    = (_Float16*)alloc(sizeof(_Float16) * Cc * Cc);
  _Float16* aI    = (_Float16*)alloc(sizeof(_Float16) * Cc * Cc);
  _Float16* aIn   = (_Float16*)alloc(sizeof(_Float16) * Cc * Cc);
  float*    zr    = (float*)alloc(sizeof(float) * (size_t)Bc * Cc * tc * HWc);
  float*    zi    = (float*)alloc(sizeof(float) * (size_t)Bc * Cc * tc * HWc);
  float*    stats = (float*)alloc(sizeof(float) * 4 * Cc);

  const int P  = Bc * Tc * HWc;        // 200704 pixels
  const int Pc = Bc * Cc * HWc;        // 1605632 (b,c,hw) threads

  corr_kernel<<<(P + 255) / 256, 256, 0, stream>>>(x, wcor, w_o);
  agg_kernel<<<(P + 255) / 256, 256, 0, stream>>>(x, w_o, x2);

  pack_xc_kernel<<<((size_t)Bc * Kp * HWc + 255) / 256, 256, 0, stream>>>(x2, w_o, Xc);
  pack_wg_kernel<<<(M1 * Kp + 255) / 256, 256, 0, stream>>>(Wg, A1);
  pack_wl_kernel<<<(Cc * Cc + 255) / 256, 256, 0, stream>>>(Wlr, Wli, aR, aI, aIn);

  gemm1_kernel<<<dim3(NT / 4, M1 / 16, Bc), 128, 0, stream>>>(A1, Xc, iterm);

  dft_mod_kernel<<<(Pc + 255) / 256, 256, 0, stream>>>(x2, iterm, w1r, w1i, ybf);

  gemm2_kernel<<<dim3(NT / 4, Cc / 16, Bc * tc), 128, 0, stream>>>(
      aR, aI, aIn, ybf, zr, zi);

  zero_stats_kernel<<<2, 256, 0, stream>>>(stats);
  stats_kernel<<<dim3(Cc, Bc, 2), 256, 0, stream>>>(zr, zi, stats);

  final_kernel<<<(Pc + 255) / 256, 256, 0, stream>>>(
      x, zr, zi, stats, gamma_r, beta_r, gamma_i, beta_i, alpha1, bias_p, out);
}
